// ContrastiveMultimodalLearning_2568390443375
// MI455X (gfx1250) — matrix operand order repaired and checked by reading, not compiled, not run on hardware
//
#include <hip/hip_runtime.h>
#include <stdint.h>

// ---------------------------------------------------------------------------
// CDNA5 (gfx1250) NT-Xent contrastive pipeline.
// All GEMMs via v_wmma_f32_16x16x32_bf16 (fp32 accum). Global->LDS staging
// uses gfx1250 async-DMA (global_load_async_to_lds_b128, ASYNCcnt) with
// double-buffered LDS. Sim matrix never materialized; deterministic sums.
// ---------------------------------------------------------------------------

typedef __attribute__((ext_vector_type(16))) __bf16 v16bf;
typedef __attribute__((ext_vector_type(8)))  float  v8f;

struct U4x2 { uint4 a, b; };

#define INV_T 14.285714285714286f  // 1/0.07

__device__ __forceinline__ unsigned short f2bf(float f) {
  unsigned int u = __float_as_uint(f);
  u += 0x7FFFu + ((u >> 16) & 1u);   // round-to-nearest-even
  return (unsigned short)(u >> 16);
}

__device__ __forceinline__ v16bf ld_frag(const unsigned short* p0,
                                         const unsigned short* p1) {
  U4x2 u;
  u.a = *(const uint4*)p0;
  u.b = *(const uint4*)p1;
  return __builtin_bit_cast(v16bf, u);
}

// gfx1250 async DMA: 16B per lane, global -> LDS, tracked by ASYNCcnt.
__device__ __forceinline__ void async_b128(unsigned ldsAddr, const void* gptr) {
  asm volatile("global_load_async_to_lds_b128 %0, %1, off"
               :: "v"(ldsAddr), "v"(gptr) : "memory");
}
__device__ __forceinline__ void wait_async0() {
  asm volatile("s_wait_asynccnt 0" ::: "memory");
}

// ---------------------------------------------------------------------------
// Core block GEMM: C[128x64] += A[128xK] * BT[64xK]^T   (A,BT bf16 row-major,
// BT is B stored N-major so fragment K-runs are contiguous in LDS).
// 256 threads = 8 waves in a 4(M) x 2(N) grid; each wave owns 32x32 via four
// 16x16 v_wmma_f32_16x16x32_bf16 accumulators. Double-buffered LDS filled by
// async DMA overlaps the copy of tile k+1 with the WMMA work of tile k.
// ---------------------------------------------------------------------------
__device__ __forceinline__ void issue_tile(const unsigned short* A,
                                           const unsigned short* BT, int K,
                                           int blockM, int blockN, int k0,
                                           unsigned aBase, unsigned bBase,
                                           int t) {
  // A tile: 128 rows x 32 bf16 = 512 x 16B; 2 async per thread
#pragma unroll
  for (int i = t; i < 512; i += 256) {
    const int r = i >> 2, c = (i & 3) << 3;
    async_b128(aBase + (unsigned)(r * 48 + c) * 2u,
               A + (size_t)(blockM + r) * K + k0 + c);
  }
  // B tile: 64 rows x 32 bf16 = 256 x 16B; 1 async per thread
  {
    const int r = t >> 2, c = (t & 3) << 3;
    async_b128(bBase + (unsigned)(r * 48 + c) * 2u,
               BT + (size_t)(blockN + r) * K + k0 + c);
  }
}

__device__ __forceinline__ void wmma_block_gemm(const unsigned short* A,
                                                const unsigned short* BT,
                                                int K, int blockM, int blockN,
                                                v8f acc[2][2]) {
  // stride 48 bf16 = 96B per row: 16B-aligned fragment reads, staggered banks
  __shared__ __align__(16) unsigned short ldsA[2][128 * 48];
  __shared__ __align__(16) unsigned short ldsB[2][64 * 48];

  const int t = threadIdx.x;
  const int lane = t & 31, wave = t >> 5;
  const int waveRow = wave & 3, waveCol = wave >> 2;
  const int hh = lane >> 4, l15 = lane & 15;

  const unsigned aB[2] = {(unsigned)(size_t)&ldsA[0][0],
                          (unsigned)(size_t)&ldsA[1][0]};
  const unsigned bB[2] = {(unsigned)(size_t)&ldsB[0][0],
                          (unsigned)(size_t)&ldsB[1][0]};

#pragma unroll
  for (int i = 0; i < 2; ++i)
#pragma unroll
    for (int j = 0; j < 2; ++j)
      acc[i][j] = v8f{0.f, 0.f, 0.f, 0.f, 0.f, 0.f, 0.f, 0.f};

  // prime buffer 0
  issue_tile(A, BT, K, blockM, blockN, 0, aB[0], bB[0], t);
  wait_async0();
  __syncthreads();

  int bsel = 0;
  for (int k0 = 0; k0 < K; k0 += 32) {
    const int nxt = bsel ^ 1;
    if (k0 + 32 < K)  // DMA tile k+1 while computing tile k
      issue_tile(A, BT, K, blockM, blockN, k0 + 32, aB[nxt], bB[nxt], t);

    const unsigned short* lA = &ldsA[bsel][0];
    const unsigned short* lB = &ldsB[bsel][0];

    // A fragment (16x32 bf16): half 0 -> K {0..7,16..23}, half 1 -> {8..15,24..31}
    v16bf afr[2], bfr[2];
#pragma unroll
    for (int tm = 0; tm < 2; ++tm) {
      const unsigned short* pa = &lA[(waveRow * 32 + tm * 16 + l15) * 48 + hh * 8];
      afr[tm] = ld_frag(pa, pa + 16);
    }
    // B fragment (32x16 bf16): half 0 -> K {0..7,16..23}, half 1 -> {8..15,24..31}
#pragma unroll
    for (int tn = 0; tn < 2; ++tn) {
      const unsigned short* pb = &lB[(waveCol * 32 + tn * 16 + l15) * 48 + hh * 8];
      bfr[tn] = ld_frag(pb, pb + 16);
    }
#pragma unroll
    for (int tm = 0; tm < 2; ++tm)
#pragma unroll
      for (int tn = 0; tn < 2; ++tn)
        acc[tm][tn] = __builtin_amdgcn_wmma_f32_16x16x32_bf16(
            false, afr[tm], false, bfr[tn], (short)0, acc[tm][tn], false, false);

    wait_async0();     // tile k+1 DMA landed (this wave)
    __syncthreads();   // ... and every wave's; also closes reads of tile k
    bsel = nxt;
  }
}

// ---------------------------------------------------------------------------
// Conversion kernels
// ---------------------------------------------------------------------------
__global__ void f2bf_kernel(const float* __restrict__ src,
                            unsigned short* __restrict__ dst, int n) {
  const int i = blockIdx.x * 256 + threadIdx.x;
  if (i < n) dst[i] = f2bf(src[i]);
}

// src [K x N] f32 row-major  ->  dst [N x K] bf16 (transposed)
__global__ void f2bf_t_kernel(const float* __restrict__ src,
                              unsigned short* __restrict__ dst, int K, int N) {
  const int i = blockIdx.x * 256 + threadIdx.x;
  if (i < K * N) {
    const int k = i / N, n = i - k * N;
    dst[(size_t)n * K + k] = f2bf(src[i]);
  }
}

// ---------------------------------------------------------------------------
// GEMM + bias (+ReLU), output bf16 or f32.  grid = (M/128, N/64), block 256.
// ---------------------------------------------------------------------------
template <int RELU, int OUTBF>
__global__ void gemm_kernel(const unsigned short* __restrict__ A,
                            const unsigned short* __restrict__ BT,
                            const float* __restrict__ bias, void* out, int N,
                            int K) {
  const int blockM = blockIdx.x * 128;
  const int blockN = blockIdx.y * 64;
  v8f acc[2][2];
  wmma_block_gemm(A, BT, K, blockM, blockN, acc);

  const int t = threadIdx.x, lane = t & 31, wave = t >> 5;
  const int waveRow = wave & 3, waveCol = wave >> 2;
  const int hh = lane >> 4, l15 = lane & 15;
#pragma unroll
  for (int tm = 0; tm < 2; ++tm)
#pragma unroll
    for (int tn = 0; tn < 2; ++tn)
#pragma unroll
      for (int r = 0; r < 8; ++r) {
        const int m = blockM + waveRow * 32 + tm * 16 + r + hh * 8;
        const int n = blockN + waveCol * 32 + tn * 16 + l15;
        float v = acc[tm][tn][r] + bias[n];
        if (RELU) v = fmaxf(v, 0.f);
        if (OUTBF)
          ((unsigned short*)out)[(size_t)m * N + n] = f2bf(v);
        else
          ((float*)out)[(size_t)m * N + n] = v;
      }
}

// ---------------------------------------------------------------------------
// Row-normalize z [8192 x 128]; write fp32 reps + bf16 reps. 1 wave / row.
// ---------------------------------------------------------------------------
__global__ void normalize_kernel(const float* __restrict__ z,
                                 float* __restrict__ repsf,
                                 unsigned short* __restrict__ repsb) {
  const int t = threadIdx.x, lane = t & 31, wave = t >> 5;
  const int row = blockIdx.x * 8 + wave;
  const float4 v = *(const float4*)(z + (size_t)row * 128 + lane * 4);
  float s = v.x * v.x + v.y * v.y + v.z * v.z + v.w * v.w;
  s += __shfl_xor(s, 1, 32);
  s += __shfl_xor(s, 2, 32);
  s += __shfl_xor(s, 4, 32);
  s += __shfl_xor(s, 8, 32);
  s += __shfl_xor(s, 16, 32);
  const float scale = 1.f / fmaxf(sqrtf(s), 1e-12f);
  float4 o;
  o.x = v.x * scale; o.y = v.y * scale; o.z = v.z * scale; o.w = v.w * scale;
  *(float4*)(repsf + (size_t)row * 128 + lane * 4) = o;
  uint2 pk;
  pk.x = (unsigned)f2bf(o.x) | ((unsigned)f2bf(o.y) << 16);
  pk.y = (unsigned)f2bf(o.z) | ((unsigned)f2bf(o.w) << 16);
  *(uint2*)(repsb + (size_t)row * 128 + lane * 4) = pk;
}

// p[i] = p[i+B] = dot(repsf[i], repsf[i+B]).  1 wave / row, rows = 4096.
__global__ void pos_kernel(const float* __restrict__ repsf,
                           float* __restrict__ p) {
  const int t = threadIdx.x, lane = t & 31, wave = t >> 5;
  const int i = blockIdx.x * 8 + wave;
  const float4 a = *(const float4*)(repsf + (size_t)i * 128 + lane * 4);
  const float4 b = *(const float4*)(repsf + (size_t)(i + 4096) * 128 + lane * 4);
  float d = a.x * b.x + a.y * b.y + a.z * b.z + a.w * b.w;
  d += __shfl_xor(d, 1, 32);
  d += __shfl_xor(d, 2, 32);
  d += __shfl_xor(d, 4, 32);
  d += __shfl_xor(d, 8, 32);
  d += __shfl_xor(d, 16, 32);
  if (lane == 0) { p[i] = d; p[i + 4096] = d; }
}

// ---------------------------------------------------------------------------
// Sim tile + exp/row-reduce epilogue.  sim never hits memory.
// partial[cb][row] = sum_{n in colblock cb, n != row} exp((s-1)/t)
// grid = (8192/128, 8192/64) = (64, 128), block 256.
// ---------------------------------------------------------------------------
__global__ void sim_kernel(const unsigned short* __restrict__ reps,
                           float* __restrict__ partial) {
  const int blockM = blockIdx.x * 128;
  const int cb = blockIdx.y;
  const int blockN = cb * 64;
  v8f acc[2][2];
  wmma_block_gemm(reps, reps, 128, blockM, blockN, acc);

  __shared__ float ldsRow[128][2];
  const int t = threadIdx.x, lane = t & 31, wave = t >> 5;
  const int waveRow = wave & 3, waveCol = wave >> 2;
  const int hh = lane >> 4, l15 = lane & 15;

#pragma unroll
  for (int tm = 0; tm < 2; ++tm)
#pragma unroll
    for (int r = 0; r < 8; ++r) {
      float rs = 0.f;
#pragma unroll
      for (int tn = 0; tn < 2; ++tn) {
        const int m = blockM + waveRow * 32 + tm * 16 + r + hh * 8;
        const int n = blockN + waveCol * 32 + tn * 16 + l15;
        const float v = acc[tm][tn][r];
        // s_ii == 1 is the row max -> exp((s-1)/t) <= 1, stable w/o max pass
        float e = (m == n) ? 0.f : __expf((v - 1.f) * INV_T);
        e += __shfl_xor(e, 1, 32);
        e += __shfl_xor(e, 2, 32);
        e += __shfl_xor(e, 4, 32);
        e += __shfl_xor(e, 8, 32);
        rs += e;  // lanes 0 / 16 now hold the 16-col sum for rows m / m+8
      }
      if (l15 == 0)
        ldsRow[waveRow * 32 + tm * 16 + r + hh * 8][waveCol] = rs;
    }
  __syncthreads();
  if (t < 128)
    partial[(size_t)cb * 8192 + blockM + t] = ldsRow[t][0] + ldsRow[t][1];
}

// loss_r = log( S_offdiag + exp((p-1)/t) ) + (1 - p)/t ; fixed-order sums.
__global__ void rowloss_kernel(const float* __restrict__ partial,
                               const float* __restrict__ p,
                               float* __restrict__ blockSums) {
  const int r = blockIdx.x * 256 + threadIdx.x;
  float S = 0.f;
  for (int cb = 0; cb < 128; ++cb) S += partial[(size_t)cb * 8192 + r];
  const float pi = p[r];
  const float tot = S + __expf((pi - 1.f) * INV_T);  // pos counted twice
  const float loss = logf(tot) + (1.f - pi) * INV_T;

  __shared__ float red[256];
  red[threadIdx.x] = loss;
  __syncthreads();
  for (int s2 = 128; s2 > 0; s2 >>= 1) {
    if (threadIdx.x < s2) red[threadIdx.x] += red[threadIdx.x + s2];
    __syncthreads();
  }
  if (threadIdx.x == 0) blockSums[blockIdx.x] = red[0];
}

__global__ void final_kernel(const float* __restrict__ blockSums,
                             float* __restrict__ out) {
  if (threadIdx.x == 0) {
    float s = 0.f;
    for (int i = 0; i < 32; ++i) s += blockSums[i];
    out[0] = s * (1.f / 8192.f);
  }
}

// ---------------------------------------------------------------------------
extern "C" void kernel_launch(void* const* d_in, const int* in_sizes, int n_in,
                              void* d_out, int out_size, void* d_ws,
                              size_t ws_size, hipStream_t stream) {
  (void)in_sizes; (void)n_in; (void)out_size; (void)ws_size;
  const float* x_img  = (const float*)d_in[0];
  const float* x_txt  = (const float*)d_in[1];
  const float* We_img = (const float*)d_in[2];
  const float* be_img = (const float*)d_in[3];
  const float* We_txt = (const float*)d_in[4];
  const float* be_txt = (const float*)d_in[5];
  const float* W1_img = (const float*)d_in[6];
  const float* b1_img = (const float*)d_in[7];
  const float* W2_img = (const float*)d_in[8];
  const float* b2_img = (const float*)d_in[9];
  const float* W1_txt = (const float*)d_in[10];
  const float* b1_txt = (const float*)d_in[11];
  const float* W2_txt = (const float*)d_in[12];
  const float* b2_txt = (const float*)d_in[13];

  constexpr int B = 4096, Din = 1024, Denc = 512, Dh = 256, Dp = 128;

  size_t off = 0;
  auto alloc = [&](size_t bytes) -> void* {
    off = (off + 255) & ~(size_t)255;
    void* p = (char*)d_ws + off;
    off += bytes;
    return p;
  };
  unsigned short* xbI  = (unsigned short*)alloc((size_t)B * Din * 2);
  unsigned short* xbT  = (unsigned short*)alloc((size_t)B * Din * 2);
  unsigned short* WetI = (unsigned short*)alloc((size_t)Denc * Din * 2);
  unsigned short* WetT = (unsigned short*)alloc((size_t)Denc * Din * 2);
  unsigned short* W1tI = (unsigned short*)alloc((size_t)Dh * Denc * 2);
  unsigned short* W1tT = (unsigned short*)alloc((size_t)Dh * Denc * 2);
  unsigned short* W2tI = (unsigned short*)alloc((size_t)Dp * Dh * 2);
  unsigned short* W2tT = (unsigned short*)alloc((size_t)Dp * Dh * 2);
  unsigned short* hI   = (unsigned short*)alloc((size_t)B * Denc * 2);
  unsigned short* hT   = (unsigned short*)alloc((size_t)B * Denc * 2);
  unsigned short* gI   = (unsigned short*)alloc((size_t)B * Dh * 2);
  unsigned short* gT   = (unsigned short*)alloc((size_t)B * Dh * 2);
  float*          zAll = (float*)alloc((size_t)2 * B * Dp * 4);
  float*          repsf= (float*)alloc((size_t)2 * B * Dp * 4);
  unsigned short* repsb= (unsigned short*)alloc((size_t)2 * B * Dp * 2);
  float*          pbuf = (float*)alloc((size_t)2 * B * 4);
  float*          part = (float*)alloc((size_t)128 * 8192 * 4);
  float*          bsum = (float*)alloc((size_t)32 * 4);

  // 1) convert activations / transpose+convert weights to bf16
  f2bf_kernel<<<(B * Din + 255) / 256, 256, 0, stream>>>(x_img, xbI, B * Din);
  f2bf_kernel<<<(B * Din + 255) / 256, 256, 0, stream>>>(x_txt, xbT, B * Din);
  f2bf_t_kernel<<<(Din * Denc + 255) / 256, 256, 0, stream>>>(We_img, WetI, Din, Denc);
  f2bf_t_kernel<<<(Din * Denc + 255) / 256, 256, 0, stream>>>(We_txt, WetT, Din, Denc);
  f2bf_t_kernel<<<(Denc * Dh + 255) / 256, 256, 0, stream>>>(W1_img, W1tI, Denc, Dh);
  f2bf_t_kernel<<<(Denc * Dh + 255) / 256, 256, 0, stream>>>(W1_txt, W1tT, Denc, Dh);
  f2bf_t_kernel<<<(Dh * Dp + 255) / 256, 256, 0, stream>>>(W2_img, W2tI, Dh, Dp);
  f2bf_t_kernel<<<(Dh * Dp + 255) / 256, 256, 0, stream>>>(W2_txt, W2tT, Dh, Dp);

  // 2) encoder: h = x @ We + be   [4096x1024]x[1024x512]
  gemm_kernel<0, 1><<<dim3(B / 128, Denc / 64), 256, 0, stream>>>(xbI, WetI, be_img, hI, Denc, Din);
  gemm_kernel<0, 1><<<dim3(B / 128, Denc / 64), 256, 0, stream>>>(xbT, WetT, be_txt, hT, Denc, Din);
  // 3) proj1: g = relu(h @ W1 + b1)   [4096x512]x[512x256]
  gemm_kernel<1, 1><<<dim3(B / 128, Dh / 64), 256, 0, stream>>>(hI, W1tI, b1_img, gI, Dh, Denc);
  gemm_kernel<1, 1><<<dim3(B / 128, Dh / 64), 256, 0, stream>>>(hT, W1tT, b1_txt, gT, Dh, Denc);
  // 4) proj2: z = g @ W2 + b2 (fp32)   [4096x256]x[256x128]; txt rows at 4096+
  gemm_kernel<0, 0><<<dim3(B / 128, Dp / 64), 256, 0, stream>>>(gI, W2tI, b2_img, zAll, Dp, Dh);
  gemm_kernel<0, 0><<<dim3(B / 128, Dp / 64), 256, 0, stream>>>(gT, W2tT, b2_txt, zAll + (size_t)B * Dp, Dp, Dh);

  // 5) normalize + positives
  normalize_kernel<<<(2 * B) / 8, 256, 0, stream>>>(zAll, repsf, repsb);
  pos_kernel<<<B / 8, 256, 0, stream>>>(repsf, pbuf);

  // 6) sim tiles -> per-colblock row partials (wmma, no sim materialization)
  sim_kernel<<<dim3(8192 / 128, 8192 / 64), 256, 0, stream>>>(repsb, part);

  // 7) deterministic logsumexp + mean
  rowloss_kernel<<<32, 256, 0, stream>>>(part, pbuf, bsum);
  final_kernel<<<1, 64, 0, stream>>>(bsum, (float*)d_out);
}
